// ScaledDotProductAttention_9354438770754
// MI455X (gfx1250) — matrix operand order repaired and checked by reading, not compiled, not run on hardware
//
#include <hip/hip_runtime.h>

typedef __attribute__((ext_vector_type(16))) _Float16 v16h;
typedef __attribute__((ext_vector_type(8)))  _Float16 v8h;
typedef __attribute__((ext_vector_type(4)))  _Float16 v4h;
typedef __attribute__((ext_vector_type(8)))  float    v8f;
typedef __attribute__((ext_vector_type(4)))  float    v4f;

#define D_MODEL 1024
#define SEQ     2048
#define BATCH   4
#define M_TOTAL (BATCH * SEQ)   // 8192 tokens

#define WMMA_F16(a, b, c) \
  __builtin_amdgcn_wmma_f32_16x16x32_f16(false, (a), false, (b), (short)0, (c), false, false)

// ---- fragment loaders (wave32, contiguous K per lane half; A and B use the same order) ----
// A (16x32 f16, MxK): lanes 0-15 -> row=lane, K 0-15 ; lanes 16-31 -> row=lane-16, K 16-31
static __device__ __forceinline__ v16h load_a_frag(const _Float16* base, int stride,
                                                   int row0, int kcol, int lane) {
  const int r  = lane & 15;
  const int k8 = (lane & 16) ? 16 : 0;
  const _Float16* p = base + (size_t)(row0 + r) * stride + kcol + k8;
  v8h lo = *(const v8h*)p;         // K = kcol+k8 .. +7
  v8h hi = *(const v8h*)(p + 8);   // K = kcol+k8+8 .. +15
  return __builtin_shufflevector(lo, hi, 0,1,2,3,4,5,6,7,8,9,10,11,12,13,14,15);
}

// B (32x16 f16, KxN) from K-contiguous storage (element (k,n) at base[n*stride + k]):
// lanes 0-15 -> col=lane, K 0-15 ; lanes 16-31 -> col=lane-16, K 16-31
static __device__ __forceinline__ v16h load_b_frag(const _Float16* base, size_t stride,
                                                   int col0, int kcol, int lane) {
  const int n   = lane & 15;
  const int k16 = (lane & 16) ? 16 : 0;
  const _Float16* p = base + (size_t)(col0 + n) * stride + kcol + k16;
  v8h lo = *(const v8h*)p;
  v8h hi = *(const v8h*)(p + 8);
  return __builtin_shufflevector(lo, hi, 0,1,2,3,4,5,6,7,8,9,10,11,12,13,14,15);
}

// =====================================================================
// Kernel 1: Q/K/V projections.  y = x @ W^T  (W is [out,in] row-major)
// Block tile 128(M) x 64(N), K-step 64, 8 waves -> each wave 32x32.
// mode 0 -> Q (row-major f16), 1 -> K (row-major f16), 2 -> V transposed f16.
// =====================================================================
__global__ __launch_bounds__(256) void qkv_proj_kernel(
    const float* __restrict__ x,
    const float* __restrict__ Wq, const float* __restrict__ Wk, const float* __restrict__ Wv,
    _Float16* __restrict__ Qh, _Float16* __restrict__ Kh, _Float16* __restrict__ Vt)
{
  constexpr int SA = 72;                 // LDS stride (halfs): 144B rows keep 16B alignment
  __shared__ _Float16 As[128 * SA];      // 18.4 KB
  __shared__ _Float16 Bs[64 * SA];       //  9.2 KB
  __shared__ _Float16 Ostage[128 * 64];  // 16.0 KB output staging

  const int tid  = threadIdx.x;
  const int lane = tid & 31;
  const int w    = tid >> 5;
  const int wm   = w >> 1;               // 0..3  -> row offset wm*32
  const int wn   = w & 1;                // 0..1  -> col offset wn*32
  const int m0   = blockIdx.y * 128;
  const int n0   = blockIdx.x * 64;
  const int mode = blockIdx.z;
  const float* W = (mode == 0) ? Wq : ((mode == 1) ? Wk : Wv);

  v8f c00 = {}, c01 = {}, c10 = {}, c11 = {};

  for (int k0 = 0; k0 < D_MODEL; k0 += 64) {
    // stage x tile 128x64 (f32 -> f16): 2048 float4 slots, 8 per thread
#pragma unroll
    for (int i = 0; i < 8; ++i) {
      int idx = tid + i * 256;
      int r = idx >> 4, c4 = (idx & 15) * 4;
      v4f f = *(const v4f*)(x + (size_t)(m0 + r) * D_MODEL + k0 + c4);
      v4h h = { (_Float16)f.x, (_Float16)f.y, (_Float16)f.z, (_Float16)f.w };
      *(v4h*)(As + r * SA + c4) = h;
    }
    // stage W tile 64x64 (f32 -> f16); Bs[n][k] = W[n0+n][k0+k]: 1024 slots, 4/thread
#pragma unroll
    for (int i = 0; i < 4; ++i) {
      int idx = tid + i * 256;
      int r = idx >> 4, c4 = (idx & 15) * 4;
      v4f f = *(const v4f*)(W + (size_t)(n0 + r) * D_MODEL + k0 + c4);
      v4h h = { (_Float16)f.x, (_Float16)f.y, (_Float16)f.z, (_Float16)f.w };
      *(v4h*)(Bs + r * SA + c4) = h;
    }
    __syncthreads();

#pragma unroll
    for (int kc = 0; kc < 64; kc += 32) {
      v16h a0 = load_a_frag(As, SA, wm * 32,      kc, lane);
      v16h a1 = load_a_frag(As, SA, wm * 32 + 16, kc, lane);
      v16h b0 = load_b_frag(Bs, SA, wn * 32,      kc, lane);
      v16h b1 = load_b_frag(Bs, SA, wn * 32 + 16, kc, lane);
      c00 = WMMA_F16(a0, b0, c00);
      c01 = WMMA_F16(a0, b1, c01);
      c10 = WMMA_F16(a1, b0, c10);
      c11 = WMMA_F16(a1, b1, c11);
    }
    __syncthreads();
  }

  // ---- epilogue: stage tile to LDS in output memory order, then b128 stores ----
  const int hi8 = (lane & 16) ? 8 : 0;
  const int cn  = lane & 15;
  const int mrow = wm * 32;              // wave row base within tile
  const int ncol = wn * 32;              // wave col base within tile

  if (mode == 2) {
    // V transposed: Ostage laid out [n][m] (64 x 128)
#pragma unroll
    for (int v = 0; v < 8; ++v) {
      Ostage[(ncol +  0 + cn) * 128 + mrow +  0 + v + hi8] = (_Float16)c00[v];
      Ostage[(ncol + 16 + cn) * 128 + mrow +  0 + v + hi8] = (_Float16)c01[v];
      Ostage[(ncol +  0 + cn) * 128 + mrow + 16 + v + hi8] = (_Float16)c10[v];
      Ostage[(ncol + 16 + cn) * 128 + mrow + 16 + v + hi8] = (_Float16)c11[v];
    }
    __syncthreads();
#pragma unroll
    for (int i = 0; i < 4; ++i) {
      int idx = tid + i * 256;           // 1024 chunks of 8 halfs
      int n = idx >> 4, c8 = (idx & 15) * 8;
      *(v8h*)(Vt + (size_t)(n0 + n) * M_TOTAL + m0 + c8) = *(const v8h*)(Ostage + n * 128 + c8);
    }
  } else {
    _Float16* __restrict__ Y = (mode == 0) ? Qh : Kh;
    // row-major: Ostage laid out [m][n] (128 x 64)
#pragma unroll
    for (int v = 0; v < 8; ++v) {
      Ostage[(mrow +  0 + v + hi8) * 64 + ncol +  0 + cn] = (_Float16)c00[v];
      Ostage[(mrow +  0 + v + hi8) * 64 + ncol + 16 + cn] = (_Float16)c01[v];
      Ostage[(mrow + 16 + v + hi8) * 64 + ncol +  0 + cn] = (_Float16)c10[v];
      Ostage[(mrow + 16 + v + hi8) * 64 + ncol + 16 + cn] = (_Float16)c11[v];
    }
    __syncthreads();
#pragma unroll
    for (int i = 0; i < 4; ++i) {
      int idx = tid + i * 256;           // 1024 chunks of 8 halfs
      int m = idx >> 3, c8 = (idx & 7) * 8;
      *(v8h*)(Y + (size_t)(m0 + m) * D_MODEL + n0 + c8) = *(const v8h*)(Ostage + m * 64 + c8);
    }
  }
}

// =====================================================================
// Kernel 2: flash attention. Block = 32 queries x full 1024 head dim.
// 8 waves: S-stage wave -> one 16x16 S tile (2x4 grid over 32x64);
// PV-stage wave -> 128-wide d-slice of O (16 accumulator tiles).
// =====================================================================
__global__ __launch_bounds__(256) void flash_attn_kernel(
    const _Float16* __restrict__ Qh, const _Float16* __restrict__ Kh,
    const _Float16* __restrict__ Vt, float* __restrict__ out)
{
  constexpr int SQ = 1032;               // Q LDS stride (halfs), 2064B rows (16B aligned)
  constexpr int SS = 72;                 // S buffer stride (floats)
  constexpr int SP = 72;                 // P buffer stride (halfs), 144B rows (16B aligned)
  __shared__ _Float16 Qs[32 * SQ];       // 66.0 KB
  __shared__ float    Sbuf[32 * SS];     //  9.2 KB
  __shared__ _Float16 Pbuf[32 * SP];     //  4.6 KB
  __shared__ float    cfac[32];
  __shared__ float    linv[32];

  const int tid  = threadIdx.x;
  const int lane = tid & 31;
  const int w    = tid >> 5;
  const int q0   = blockIdx.x * 32;
  const int b    = blockIdx.y;
  const size_t qrow0 = (size_t)b * SEQ + q0;

  // pin the 32x1024 Q tile in LDS via CDNA5 async global->LDS DMA (ASYNCcnt)
#pragma unroll
  for (int i = 0; i < 16; ++i) {
    int idx = tid + i * 256;             // 4096 chunks of 8 halfs
    int r = idx >> 7, c8 = (idx & 127) * 8;
    const _Float16* src = Qh + (qrow0 + r) * D_MODEL + c8;
    unsigned lds_dst = (unsigned)(size_t)(Qs + r * SQ + c8);
    asm volatile("global_load_async_to_lds_b128 %0, %1, off"
                 :: "v"(lds_dst), "v"(src) : "memory");
  }
  asm volatile("s_wait_asynccnt 0" ::: "memory");

  float m_run = -3.0e38f, l_run = 0.0f;  // per-row state, valid in wave 0 (tid<32)
  v8f o[16];
#pragma unroll
  for (int i = 0; i < 16; ++i) o[i] = (v8f){};

  __syncthreads();

  const int sm  = (w >> 2) * 16;         // S tile row
  const int sn  = (w & 3) * 16;          // S tile col
  const int d0  = w * 128;               // this wave's O d-slice
  const int hi8 = (lane & 16) ? 8 : 0;
  const int cn  = lane & 15;
  const int k16 = (lane & 16) ? 16 : 0;

  for (int it = 0; it < SEQ / 64; ++it) {
    const int kv0 = it * 64;

    // ---------- S = Q K^T (one 16x16 tile per wave, d = 1024) ----------
    v8f s = (v8f){};
    const _Float16* kcol = Kh + (size_t)(b * SEQ + kv0 + sn + cn) * D_MODEL + k16;
#pragma unroll 4
    for (int d = 0; d < D_MODEL; d += 32) {
      v16h a  = load_a_frag(Qs, SQ, sm, d, lane);
      v8h blo = *(const v8h*)(kcol + d);
      v8h bhi = *(const v8h*)(kcol + d + 8);
      v16h bb = __builtin_shufflevector(blo, bhi, 0,1,2,3,4,5,6,7,8,9,10,11,12,13,14,15);
      s = WMMA_F16(a, bb, s);
    }
#pragma unroll
    for (int v = 0; v < 8; ++v)
      Sbuf[(sm + v + hi8) * SS + sn + cn] = s[v];
    __syncthreads();

    // ---------- online softmax: one thread per query row ----------
    if (tid < 32) {
      float r[64];
      float mx = m_run;
#pragma unroll
      for (int c = 0; c < 64; ++c) {
        r[c] = Sbuf[tid * SS + c] * 0.03125f;   // 1/sqrt(1024)
        mx = fmaxf(mx, r[c]);
      }
      float cf  = __expf(m_run - mx);
      float sum = 0.0f;
#pragma unroll
      for (int c = 0; c < 64; ++c) {
        float p = __expf(r[c] - mx);
        sum += p;
        Pbuf[tid * SP + c] = (_Float16)p;
      }
      l_run = l_run * cf + sum;
      m_run = mx;
      cfac[tid] = cf;
    }
    __syncthreads();

    // ---------- O = diag(cf) * O + P V ----------
#pragma unroll
    for (int tm = 0; tm < 2; ++tm)
#pragma unroll
      for (int v = 0; v < 8; ++v) {
        float cfv = cfac[tm * 16 + v + hi8];
#pragma unroll
        for (int j = 0; j < 8; ++j) o[tm * 8 + j][v] *= cfv;
      }

    v16h pa[2][2];
#pragma unroll
    for (int tm = 0; tm < 2; ++tm)
#pragma unroll
      for (int ks = 0; ks < 2; ++ks)
        pa[tm][ks] = load_a_frag(Pbuf, SP, tm * 16, ks * 32, lane);

#pragma unroll
    for (int j = 0; j < 8; ++j) {
      const _Float16* vcol = Vt + (size_t)(d0 + j * 16 + cn) * M_TOTAL + (size_t)b * SEQ + kv0 + k16;
#pragma unroll
      for (int ks = 0; ks < 2; ++ks) {
        v8h blo = *(const v8h*)(vcol + ks * 32);
        v8h bhi = *(const v8h*)(vcol + ks * 32 + 8);
        v16h bb = __builtin_shufflevector(blo, bhi, 0,1,2,3,4,5,6,7,8,9,10,11,12,13,14,15);
        o[0 * 8 + j] = WMMA_F16(pa[0][ks], bb, o[0 * 8 + j]);
        o[1 * 8 + j] = WMMA_F16(pa[1][ks], bb, o[1 * 8 + j]);
      }
    }
    __syncthreads();
  }

  if (tid < 32) linv[tid] = 1.0f / l_run;
  __syncthreads();

#pragma unroll
  for (int tm = 0; tm < 2; ++tm)
#pragma unroll
    for (int v = 0; v < 8; ++v) {
      int row = tm * 16 + v + hi8;
      float sc = linv[row];
#pragma unroll
      for (int j = 0; j < 8; ++j)
        out[(qrow0 + row) * D_MODEL + d0 + j * 16 + cn] = o[tm * 8 + j][v] * sc;
    }
}

// =====================================================================
extern "C" void kernel_launch(void* const* d_in, const int* in_sizes, int n_in,
                              void* d_out, int out_size, void* d_ws, size_t ws_size,
                              hipStream_t stream) {
  const float* x  = (const float*)d_in[0];
  const float* Wq = (const float*)d_in[1];
  const float* Wk = (const float*)d_in[2];
  const float* Wv = (const float*)d_in[3];
  float* out = (float*)d_out;

  // workspace: Q (16MB f16) | K (16MB f16) | V^T (16MB f16)
  _Float16* Qh = (_Float16*)d_ws;
  _Float16* Kh = Qh + (size_t)M_TOTAL * D_MODEL;
  _Float16* Vt = Kh + (size_t)M_TOTAL * D_MODEL;

  qkv_proj_kernel<<<dim3(D_MODEL / 64, M_TOTAL / 128, 3), 256, 0, stream>>>(
      x, Wq, Wk, Wv, Qh, Kh, Vt);
  flash_attn_kernel<<<dim3(SEQ / 32, BATCH), 256, 0, stream>>>(Qh, Kh, Vt, out);
}
